// VQVAE_21303037788262
// MI455X (gfx1250) — matrix operand order repaired and checked by reading, not compiled, not run on hardware
//
#include <hip/hip_runtime.h>
#include <hip/hip_bf16.h>
#include <cstdint>
#include <cstddef>

typedef __bf16 bf16_t;
typedef __attribute__((ext_vector_type(16))) __bf16 v16bf;
typedef __attribute__((ext_vector_type(8)))  float  v8f;

// ---------------- problem dimensions ----------------
#define BATCH 32
#define LX    8192     // input length
#define LP1   2046     // after conv1(k=8)+pool(5,4)
#define L2C   2039     // enc2 conv out
#define L2P   509      // after pool(4,4)
#define L3C   502      // enc3 conv out
#define TQ    251      // latent time
#define LU2   502      // upsample x2
#define LD3   510      // dec3 convT out
#define LU3   2040     // upsample x4
#define LD2   2048     // dec2 convT out
#define LU1   8192     // upsample x4
#define LDEC  8200     // final convT out
#define NCODE 200
#define DCODE 8

// padded activation geometry (margins hold zeros, cleared by memset)
#define PADL  16
#define LP1P  (LP1 + 96)    // 2142
#define L2PP  (L2P + 96)    // 605
#define TQP   (TQ  + 96)    // 347
#define LU2P  (LU2 + 96)    // 598
#define LU3P  (LU3 + 96)    // 2136
#define LU1P  (LU1 + 96)    // 8288
#define CP9   303           // ceil(2720/9): channel-padded Cin for K=9, Cin=300

// =====================================================================
// Encoder branches: conv1d(k=8) + relu + maxpool(5,4) -> fp32 [B][256][LP1]
// =====================================================================
__global__ __launch_bounds__(256) void encoder_branch_kernel(
    const float* __restrict__ x,
    const float* __restrict__ w1,  const float* __restrict__ b1,
    const float* __restrict__ w11, const float* __restrict__ b11,
    const float* __restrict__ w12, const float* __restrict__ b12,
    float* __restrict__ preLN)
{
    int idx = blockIdx.x * blockDim.x + threadIdx.x;
    const int total = BATCH * 256 * LP1;
    if (idx >= total) return;
    int t  = idx % LP1;
    int ch = (idx / LP1) % 256;
    int b  = idx / (LP1 * 256);
    const float* xb = x + (size_t)b * 6 * LX;
    float best = -3.4e38f;
    for (int j = 0; j < 5; ++j) {
        int p = 4 * t + j;
        float s;
        if (ch < 128) {
            s = b1[ch];
            #pragma unroll
            for (int ic = 0; ic < 4; ++ic) {
                const float* xp = xb + (size_t)ic * LX + p;
                const float* wp = w1 + (ch * 4 + ic) * 8;
                #pragma unroll
                for (int kk = 0; kk < 8; ++kk) s += wp[kk] * xp[kk];
            }
        } else if (ch < 192) {
            int c = ch - 128;
            s = b11[c];
            const float* xp = xb + (size_t)4 * LX + p;
            const float* wp = w11 + c * 8;
            #pragma unroll
            for (int kk = 0; kk < 8; ++kk) s += wp[kk] * xp[kk];
        } else {
            int c = ch - 192;
            s = b12[c];
            const float* xp = xb + (size_t)5 * LX + p;
            const float* wp = w12 + c * 8;
            #pragma unroll
            for (int kk = 0; kk < 8; ++kk) s += wp[kk] * xp[kk];
        }
        best = fmaxf(best, s);
    }
    preLN[idx] = fmaxf(best, 0.f);
}

// =====================================================================
// Per-(b,t) segmented channel LayerNorm -> padded bf16 [B][256][LP1P]
// =====================================================================
__global__ __launch_bounds__(256) void ln_kernel(
    const float* __restrict__ preLN,
    const float* __restrict__ g1,  const float* __restrict__ be1,
    const float* __restrict__ g11, const float* __restrict__ be11,
    const float* __restrict__ g12, const float* __restrict__ be12,
    bf16_t* __restrict__ out)
{
    int bt = blockIdx.x;
    int t = bt % LP1, b = bt / LP1;
    int ch = threadIdx.x;
    float v = preLN[((size_t)b * 256 + ch) * LP1 + t];
    __shared__ float ssum[256], ssq[256];
    ssum[ch] = v; ssq[ch] = v * v;
    __syncthreads();
    int off, S;
    if (ch < 128)      { off = 0;   S = 128; }
    else if (ch < 192) { off = 128; S = 64;  }
    else               { off = 192; S = 64;  }
    int l = ch - off;
    for (int stride = 64; stride > 0; stride >>= 1) {
        if (l < stride && l + stride < S) {
            ssum[ch] += ssum[ch + stride];
            ssq[ch]  += ssq[ch + stride];
        }
        __syncthreads();
    }
    float mean = ssum[off] / (float)S;
    float var  = ssq[off] / (float)S - mean * mean;
    float scale = rsqrtf(var + 1e-5f);
    float g, bb;
    if (ch < 128)      { g = g1[ch];        bb = be1[ch]; }
    else if (ch < 192) { g = g11[ch - 128]; bb = be11[ch - 128]; }
    else               { g = g12[ch - 192]; bb = be12[ch - 192]; }
    out[((size_t)b * 256 + ch) * LP1P + t + PADL] = (bf16_t)((v - mean) * scale * g + bb);
}

// =====================================================================
// WMMA conv1d: one wave32 computes a (MT*16)(M=oc) x 64(N=t) output tile.
// MT A-fragments reuse the same 4 B-fragments each K-step -> 4*MT WMMAs
// per 64 B-gathers. Inputs padded: K-loop fully branch-free.
// =====================================================================
union AFrag { v16bf v; uint4 q[2]; };

template <int KSZ, int ACT, int OUTBF16, int MT>
__global__ __launch_bounds__(32) void conv1d_wmma(
    const bf16_t* __restrict__ X, const bf16_t* __restrict__ W,
    const float* __restrict__ bias,
    float* __restrict__ Yf, bf16_t* __restrict__ Yh,
    int LinP, int XbStride, int Cout, int Lout,
    int tbase /* PADL - pad */, int kend, int Ystride, int Ytoff)
{
    const int lane = threadIdx.x & 31;
    const int grp  = lane >> 4;
    const int nidx = lane & 15;
    const int t0 = blockIdx.x * 64;
    const int m0 = blockIdx.y * (16 * MT);
    const int b  = blockIdx.z;

    const bf16_t* Xb = X + (size_t)b * XbStride;
    const bf16_t* Wbase = W + (size_t)(m0 + nidx) * kend;  // rows zero-padded
    const int tcol = t0 + nidx + tbase;                    // always in-bounds
    const int grp8  = grp * 8;
    const int grp16 = grp * 16;

    v8f acc[MT][4] = {};
    for (int k0 = 0; k0 < kend; k0 += 32) {
        __builtin_prefetch((const void*)(Wbase + k0 + 128), 0, 3);
        // A fragments: per lane two contiguous 16B runs per M-subtile
        AFrag a[MT];
        #pragma unroll
        for (int mt = 0; mt < MT; ++mt) {
            const bf16_t* wr = Wbase + (size_t)(mt * 16) * kend + k0 + grp16;
            a[mt].q[0] = *(const uint4*)wr;
            a[mt].q[1] = *(const uint4*)(wr + 8);
        }
        // B fragments for 4 N-subtiles (shared address + imm offsets)
        v16bf bf0, bf1, bf2, bf3;
        #pragma unroll
        for (int idx = 0; idx < 16; ++idx) {
            int kk  = k0 + grp16 + idx;
            int ic  = kk / KSZ;
            int kap = kk - ic * KSZ;
            const bf16_t* p = Xb + (size_t)ic * LinP + tcol + kap;
            bf0[idx] = p[0];
            bf1[idx] = p[16];
            bf2[idx] = p[32];
            bf3[idx] = p[48];
        }
        #pragma unroll
        for (int mt = 0; mt < MT; ++mt) {
            acc[mt][0] = __builtin_amdgcn_wmma_f32_16x16x32_bf16(false, a[mt].v, false, bf0, (short)0, acc[mt][0], false, false);
            acc[mt][1] = __builtin_amdgcn_wmma_f32_16x16x32_bf16(false, a[mt].v, false, bf1, (short)0, acc[mt][1], false, false);
            acc[mt][2] = __builtin_amdgcn_wmma_f32_16x16x32_bf16(false, a[mt].v, false, bf2, (short)0, acc[mt][2], false, false);
            acc[mt][3] = __builtin_amdgcn_wmma_f32_16x16x32_bf16(false, a[mt].v, false, bf3, (short)0, acc[mt][3], false, false);
        }
    }
    // store D: VGPR r -> M = r + 8*grp, N = lane&15
    #pragma unroll
    for (int mt = 0; mt < MT; ++mt) {
        #pragma unroll
        for (int s = 0; s < 4; ++s) {
            int t = t0 + s * 16 + nidx;
            if (t < Lout) {
                #pragma unroll
                for (int r = 0; r < 8; ++r) {
                    int m = m0 + mt * 16 + grp8 + r;
                    if (m < Cout) {
                        float v = acc[mt][s][r] + bias[m];
                        if (ACT == 1) v = v > 0.f ? v : 0.f;
                        size_t o = ((size_t)b * Cout + m) * Ystride + Ytoff + t;
                        if (OUTBF16) Yh[o] = (bf16_t)v;
                        else         Yf[o] = v;
                    }
                }
            }
        }
    }
}

// =====================================================================
// maxpool1d fp32 -> padded bf16
// =====================================================================
__global__ __launch_bounds__(256) void pool_bf16_kernel(
    const float* __restrict__ X, bf16_t* __restrict__ Y,
    int C, int Lin, int Lout, int LoutP, int k, int s, int total)
{
    int idx = blockIdx.x * blockDim.x + threadIdx.x;
    if (idx >= total) return;
    int t = idx % Lout;
    int c = (idx / Lout) % C;
    int b = idx / (Lout * C);
    const float* xp = X + ((size_t)b * C + c) * Lin + (size_t)t * s;
    float m = -3.4e38f;
    for (int j = 0; j < k; ++j) m = fmaxf(m, xp[j]);
    Y[((size_t)b * C + c) * LoutP + t + PADL] = (bf16_t)m;
}

// =====================================================================
// VQ (eval)
// =====================================================================
__global__ __launch_bounds__(256) void vq_kernel(
    const float* __restrict__ z_e,       // [B][128][TQ]
    const float* __restrict__ codebook,  // [200][8]
    float* __restrict__ quant_out,       // d_out region 0 [B][128][TQ]
    bf16_t* __restrict__ quant_bf,       // plain [B][128][TQ]
    float* __restrict__ index_out,       // d_out region 2 [B][TQ][16]
    int* __restrict__ counts, float* __restrict__ sse)
{
    __shared__ float cb[NCODE * DCODE];
    for (int i = threadIdx.x; i < NCODE * DCODE; i += 256) cb[i] = codebook[i];
    __syncthreads();

    const int N = BATCH * TQ * 16;
    int r = blockIdx.x * blockDim.x + threadIdx.x;
    float localsse = 0.f;
    if (r < N) {
        int c8 = r % 16;
        int t  = (r / 16) % TQ;
        int b  = r / (16 * TQ);
        size_t base = ((size_t)b * 128 + c8 * 8) * TQ + t;
        float f[DCODE];
        #pragma unroll
        for (int d = 0; d < DCODE; ++d) f[d] = z_e[base + (size_t)d * TQ];
        int best = 0; float bd = 3.4e38f;
        for (int k = 0; k < NCODE; ++k) {
            float dist = 0.f;
            #pragma unroll
            for (int d = 0; d < DCODE; ++d) {
                float diff = f[d] - cb[k * DCODE + d];
                dist += diff * diff;
            }
            if (dist < bd) { bd = dist; best = k; }
        }
        #pragma unroll
        for (int d = 0; d < DCODE; ++d) {
            float q = cb[best * DCODE + d];
            size_t o = base + (size_t)d * TQ;
            quant_out[o] = q;
            quant_bf[o]  = (bf16_t)q;
            float diff = q - f[d];
            localsse += diff * diff;
        }
        index_out[((size_t)b * TQ + t) * 16 + c8] = (float)best;
        atomicAdd(&counts[best], 1);
    }
    __shared__ float red[256];
    red[threadIdx.x] = localsse;
    __syncthreads();
    for (int st = 128; st > 0; st >>= 1) {
        if (threadIdx.x < st) red[threadIdx.x] += red[threadIdx.x + st];
        __syncthreads();
    }
    if (threadIdx.x == 0) atomicAdd(sse, red[0]);
}

__global__ __launch_bounds__(256) void vq_finalize_kernel(
    const int* __restrict__ counts, const float* __restrict__ sse,
    float* __restrict__ out2)
{
    const int N = BATCH * TQ * 16;
    __shared__ float red[256];
    float acc = 0.f;
    for (int k = threadIdx.x; k < NCODE; k += 256) {
        float avg = (float)counts[k] / (float)N;
        acc += avg * logf(avg + 1e-10f);
    }
    red[threadIdx.x] = acc;
    __syncthreads();
    for (int st = 128; st > 0; st >>= 1) {
        if (threadIdx.x < st) red[threadIdx.x] += red[threadIdx.x + st];
        __syncthreads();
    }
    if (threadIdx.x == 0) {
        out2[0] = 0.25f * (sse[0] / (float)(N * DCODE));
        out2[1] = expf(-red[0]);
    }
}

// =====================================================================
// linear upsample (align_corners=False) -> padded bf16
// =====================================================================
template <typename TIN>
__global__ __launch_bounds__(256) void upsample_kernel(
    const TIN* __restrict__ X, bf16_t* __restrict__ Y,
    int C, int Lin, int S, int CP, int LoutP, int total)
{
    int idx = blockIdx.x * blockDim.x + threadIdx.x;
    if (idx >= total) return;
    int Lout = Lin * S;
    int j = idx % Lout;
    int c = (idx / Lout) % C;
    int b = idx / (Lout * C);
    float coord = ((float)j + 0.5f) / (float)S - 0.5f;
    coord = fminf(fmaxf(coord, 0.f), (float)(Lin - 1));
    int i0 = (int)floorf(coord);
    int i1 = (i0 + 1 < Lin) ? i0 + 1 : Lin - 1;
    float w = coord - (float)i0;
    const TIN* xp = X + ((size_t)b * C + c) * Lin;
    float v0 = (float)xp[i0], v1 = (float)xp[i1];
    Y[((size_t)b * CP + c) * LoutP + j + PADL] = (bf16_t)(v0 * (1.f - w) + v1 * w);
}

// =====================================================================
// weight prep (zero-padded [Mpad][kend] bf16)
// =====================================================================
__global__ __launch_bounds__(256) void cast_bf16_kernel(
    const float* __restrict__ src, bf16_t* __restrict__ dst, int n)
{
    int i = blockIdx.x * blockDim.x + threadIdx.x;
    if (i < n) dst[i] = (bf16_t)src[i];
}

// src: ConvTranspose1d weight [I][O][K] -> dst [Mpad][kend], flipped/transposed
__global__ __launch_bounds__(256) void prep_convT_pad_kernel(
    const float* __restrict__ src, bf16_t* __restrict__ dst,
    int I, int O, int K, int Mpad, int kend)
{
    int idx = blockIdx.x * blockDim.x + threadIdx.x;
    int total = Mpad * kend;
    if (idx >= total) return;
    int m = idx / kend;
    int k = idx % kend;
    float v = 0.f;
    if (m < O && k < I * K) {
        int i = k / K, kap = k % K;
        v = src[((size_t)i * O + m) * K + (K - 1 - kap)];
    }
    dst[idx] = (bf16_t)v;
}

// pack final convT trio into [16][2720] (6 valid rows) + bias6
__global__ __launch_bounds__(256) void pack_ct_kernel(
    const float* __restrict__ ct1w, const float* __restrict__ ct11w,
    const float* __restrict__ ct12w,
    const float* __restrict__ ct1b, const float* __restrict__ ct11b,
    const float* __restrict__ ct12b,
    bf16_t* __restrict__ W, float* __restrict__ bias6)
{
    int idx = blockIdx.x * blockDim.x + threadIdx.x;
    const int total = 16 * 2720;
    if (idx >= total) return;
    int m = idx / 2720;
    int k = idx % 2720;
    float v = 0.f;
    if (m < 6 && k < 2700) {
        int i = k / 9, kap = k % 9;
        if (m < 4)       v = ct1w[((size_t)i * 4 + m) * 9 + (8 - kap)];
        else if (m == 4) v = ct11w[(size_t)i * 9 + (8 - kap)];
        else             v = ct12w[(size_t)i * 9 + (8 - kap)];
    }
    W[idx] = (bf16_t)v;
    if (idx < 6) bias6[idx] = (idx < 4) ? ct1b[idx] : (idx == 4 ? ct11b[0] : ct12b[0]);
}

// =====================================================================
// final postprocess: BN(eval)+sigmoid ch0-3, relu ch4-5
// =====================================================================
__global__ __launch_bounds__(256) void postproc_kernel(
    const float* __restrict__ dfin,
    const float* __restrict__ bn_g, const float* __restrict__ bn_b,
    const float* __restrict__ bn_rm, const float* __restrict__ bn_rv,
    float* __restrict__ out)
{
    int idx = blockIdx.x * blockDim.x + threadIdx.x;
    const int total = BATCH * 6 * LDEC;
    if (idx >= total) return;
    int ch = (idx / LDEC) % 6;
    float v = dfin[idx];
    if (ch < 4) {
        v = (v - bn_rm[ch]) * rsqrtf(bn_rv[ch] + 1e-5f) * bn_g[ch] + bn_b[ch];
        v = 1.f / (1.f + expf(-v));
    } else {
        v = fmaxf(v, 0.f);
    }
    out[idx] = v;
}

// =====================================================================
// host driver
// =====================================================================
static inline int cdiv(int a, int b) { return (a + b - 1) / b; }

extern "C" void kernel_launch(void* const* d_in, const int* in_sizes, int n_in,
                              void* d_out, int out_size, void* d_ws, size_t ws_size,
                              hipStream_t stream)
{
    (void)in_sizes; (void)n_in; (void)out_size; (void)ws_size;
    const float* x       = (const float*)d_in[0];
    const float* conv1_w = (const float*)d_in[1];
    const float* conv1_b = (const float*)d_in[2];
    const float* ln1_g   = (const float*)d_in[3];
    const float* ln1_b   = (const float*)d_in[4];
    const float* conv11_w= (const float*)d_in[5];
    const float* conv11_b= (const float*)d_in[6];
    const float* ln11_g  = (const float*)d_in[7];
    const float* ln11_b  = (const float*)d_in[8];
    const float* conv12_w= (const float*)d_in[9];
    const float* conv12_b= (const float*)d_in[10];
    const float* ln12_g  = (const float*)d_in[11];
    const float* ln12_b  = (const float*)d_in[12];
    const float* convf_w = (const float*)d_in[13];
    const float* convf_b = (const float*)d_in[14];
    const float* enc2_w  = (const float*)d_in[15];
    const float* enc2_b  = (const float*)d_in[16];
    const float* enc3_w  = (const float*)d_in[17];
    const float* enc3_b  = (const float*)d_in[18];
    const float* qc_w    = (const float*)d_in[19];
    const float* qc_b    = (const float*)d_in[20];
    const float* codebook= (const float*)d_in[21];
    const float* dec3_w  = (const float*)d_in[22];
    const float* dec3_b  = (const float*)d_in[23];
    const float* dec2_w  = (const float*)d_in[24];
    const float* dec2_b  = (const float*)d_in[25];
    const float* ct1_w   = (const float*)d_in[26];
    const float* ct1_b   = (const float*)d_in[27];
    const float* bn_g    = (const float*)d_in[28];
    const float* bn_b    = (const float*)d_in[29];
    const float* bn_rm   = (const float*)d_in[30];
    const float* bn_rv   = (const float*)d_in[31];
    const float* ct11_w  = (const float*)d_in[32];
    const float* ct11_b  = (const float*)d_in[33];
    const float* ct12_w  = (const float*)d_in[34];
    const float* ct12_b  = (const float*)d_in[35];

    char* ws = (char*)d_ws;
    size_t off = 0;
    auto take = [&](size_t bytes) -> char* {
        char* p = ws + off;
        off += (bytes + 255) & ~(size_t)255;
        return p;
    };
    // fp32 scratch (aliased lifetimes): preLN -> enc2 conv out -> dec2 conv out
    float*  wsA  = (float*) take((size_t)BATCH * 300 * LD2 * 4);
    bf16_t* cat  = (bf16_t*)take((size_t)BATCH * 256 * LP1P * 2);
    bf16_t* enc1 = (bf16_t*)take((size_t)BATCH * 256 * LP1P * 2);
    bf16_t* e2p  = (bf16_t*)take((size_t)BATCH * 256 * L2PP * 2);
    // fp32 scratch: enc3 conv out -> dec3 conv out
    float*  wsE  = (float*) take((size_t)BATCH * 300 * LD3 * 4);
    bf16_t* e3p  = (bf16_t*)take((size_t)BATCH * 128 * TQP * 2);
    float*  ze   = (float*) take((size_t)BATCH * 128 * TQ  * 4);
    bf16_t* qbf  = (bf16_t*)take((size_t)BATCH * 128 * TQ  * 2);
    bf16_t* up2  = (bf16_t*)take((size_t)BATCH * 128 * LU2P * 2);
    bf16_t* up3  = (bf16_t*)take((size_t)BATCH * CP9 * LU3P * 2);
    bf16_t* ubuf = (bf16_t*)take((size_t)BATCH * CP9 * LU1P * 2);
    float*  dfin = (float*) take((size_t)BATCH * 6   * LDEC * 4);
    bf16_t* wf   = (bf16_t*)take((size_t)256 * 256  * 2);
    bf16_t* we2  = (bf16_t*)take((size_t)256 * 2048 * 2);
    bf16_t* we3  = (bf16_t*)take((size_t)128 * 2048 * 2);
    bf16_t* wq   = (bf16_t*)take((size_t)128 * 128  * 2);
    bf16_t* wd3  = (bf16_t*)take((size_t)320 * 1152 * 2);   // Mpad=320 (MT=2)
    bf16_t* wd2  = (bf16_t*)take((size_t)320 * 2720 * 2);   // Mpad=320 (MT=2)
    bf16_t* wct  = (bf16_t*)take((size_t)16  * 2720 * 2);
    float*  bias6= (float*) take(6 * 4);
    int*    stats= (int*)   take(256 * 4);
    int*    counts = stats;
    float*  sse    = (float*)(stats + NCODE);

    float* out_quant = (float*)d_out;
    float* out_dec   = out_quant + (size_t)BATCH * 128 * TQ;
    float* out_idx   = out_dec   + (size_t)BATCH * 6 * LDEC;
    float* out_stats = out_idx   + (size_t)BATCH * TQ * 16;

    // zero stats + all padded-activation margins
    hipMemsetAsync(stats, 0, 256 * 4, stream);
    hipMemsetAsync(cat,  0, (size_t)BATCH * 256 * LP1P * 2, stream);
    hipMemsetAsync(enc1, 0, (size_t)BATCH * 256 * LP1P * 2, stream);
    hipMemsetAsync(e2p,  0, (size_t)BATCH * 256 * L2PP * 2, stream);
    hipMemsetAsync(e3p,  0, (size_t)BATCH * 128 * TQP * 2, stream);
    hipMemsetAsync(up2,  0, (size_t)BATCH * 128 * LU2P * 2, stream);
    hipMemsetAsync(up3,  0, (size_t)BATCH * CP9 * LU3P * 2, stream);
    hipMemsetAsync(ubuf, 0, (size_t)BATCH * CP9 * LU1P * 2, stream);

    // ---- weight prep ----
    cast_bf16_kernel<<<cdiv(256*256, 256), 256, 0, stream>>>(convf_w, wf, 256*256);
    cast_bf16_kernel<<<cdiv(256*2048,256), 256, 0, stream>>>(enc2_w, we2, 256*2048);
    cast_bf16_kernel<<<cdiv(128*2048,256), 256, 0, stream>>>(enc3_w, we3, 128*2048);
    cast_bf16_kernel<<<cdiv(128*128, 256), 256, 0, stream>>>(qc_w,   wq,  128*128);
    prep_convT_pad_kernel<<<cdiv(320*1152,256), 256, 0, stream>>>(dec3_w, wd3, 128, 300, 9, 320, 1152);
    prep_convT_pad_kernel<<<cdiv(320*2720,256), 256, 0, stream>>>(dec2_w, wd2, 300, 300, 9, 320, 2720);
    pack_ct_kernel<<<cdiv(16*2720,256), 256, 0, stream>>>(ct1_w, ct11_w, ct12_w,
                                                          ct1_b, ct11_b, ct12_b, wct, bias6);

    // ---- encoder ----
    encoder_branch_kernel<<<cdiv(BATCH*256*LP1, 256), 256, 0, stream>>>(
        x, conv1_w, conv1_b, conv11_w, conv11_b, conv12_w, conv12_b, wsA);
    ln_kernel<<<BATCH * LP1, 256, 0, stream>>>(
        wsA, ln1_g, ln1_b, ln11_g, ln11_b, ln12_g, ln12_b, cat);

    dim3 wblk(32, 1, 1);
    // convf 1x1 256->256 relu -> padded bf16 enc1   (MT=2: 8 M-blocks)
    conv1d_wmma<1, 1, 1, 2><<<dim3(cdiv(LP1,64), 8, BATCH), wblk, 0, stream>>>(
        cat, wf, convf_b, nullptr, enc1,
        LP1P, 256*LP1P, 256, LP1, PADL, 256, LP1P, PADL);
    // enc2 k=8 256->256 relu -> fp32
    conv1d_wmma<8, 1, 0, 2><<<dim3(cdiv(L2C,64), 8, BATCH), wblk, 0, stream>>>(
        enc1, we2, enc2_b, wsA, nullptr,
        LP1P, 256*LP1P, 256, L2C, PADL, 2048, L2C, 0);
    pool_bf16_kernel<<<cdiv(BATCH*256*L2P, 256), 256, 0, stream>>>(
        wsA, e2p, 256, L2C, L2P, L2PP, 4, 4, BATCH*256*L2P);
    // enc3 k=8 256->128 relu -> fp32
    conv1d_wmma<8, 1, 0, 2><<<dim3(cdiv(L3C,64), 4, BATCH), wblk, 0, stream>>>(
        e2p, we3, enc3_b, wsE, nullptr,
        L2PP, 256*L2PP, 128, L3C, PADL, 2048, L3C, 0);
    pool_bf16_kernel<<<cdiv(BATCH*128*TQ, 256), 256, 0, stream>>>(
        wsE, e3p, 128, L3C, TQ, TQP, 2, 2, BATCH*128*TQ);
    // qc 1x1 128->128 -> fp32 z_e
    conv1d_wmma<1, 0, 0, 2><<<dim3(cdiv(TQ,64), 4, BATCH), wblk, 0, stream>>>(
        e3p, wq, qc_b, ze, nullptr,
        TQP, 128*TQP, 128, TQ, PADL, 128, TQ, 0);

    // ---- VQ ----
    vq_kernel<<<cdiv(BATCH*TQ*16, 256), 256, 0, stream>>>(
        ze, codebook, out_quant, qbf, out_idx, counts, sse);
    vq_finalize_kernel<<<1, 256, 0, stream>>>(counts, sse, out_stats);

    // ---- decoder ----
    upsample_kernel<bf16_t><<<cdiv(BATCH*128*LU2, 256), 256, 0, stream>>>(
        qbf, up2, 128, TQ, 2, 128, LU2P, BATCH*128*LU2);
    // dec3 convT k=9 pad=8, 128->300 relu -> fp32   (Mpad=320 -> 10 M-blocks)
    conv1d_wmma<9, 1, 0, 2><<<dim3(cdiv(LD3,64), 10, BATCH), wblk, 0, stream>>>(
        up2, wd3, dec3_b, wsE, nullptr,
        LU2P, 128*LU2P, 300, LD3, PADL-8, 1152, LD3, 0);
    upsample_kernel<float><<<cdiv(BATCH*300*LU3, 256), 256, 0, stream>>>(
        wsE, up3, 300, LD3, 4, CP9, LU3P, BATCH*300*LU3);
    // dec2 convT k=9 pad=8, 300->300 relu -> fp32
    conv1d_wmma<9, 1, 0, 2><<<dim3(cdiv(LD2,64), 10, BATCH), wblk, 0, stream>>>(
        up3, wd2, dec2_b, wsA, nullptr,
        LU3P, CP9*LU3P, 300, LD2, PADL-8, 2720, LD2, 0);
    upsample_kernel<float><<<cdiv(BATCH*300*LU1, 256), 256, 0, stream>>>(
        wsA, ubuf, 300, LD2, 4, CP9, LU1P, BATCH*300*LU1);
    // final convT trio k=9 pad=8, 300->6 -> fp32 (MT=1, 6 valid rows)
    conv1d_wmma<9, 0, 0, 1><<<dim3(cdiv(LDEC,64), 1, BATCH), wblk, 0, stream>>>(
        ubuf, wct, bias6, dfin, nullptr,
        LU1P, CP9*LU1P, 6, LDEC, PADL-8, 2720, LDEC, 0);
    postproc_kernel<<<cdiv(BATCH*6*LDEC, 256), 256, 0, stream>>>(
        dfin, bn_g, bn_b, bn_rm, bn_rv, out_dec);
}
